// GPRGNN_51565377356342
// MI455X (gfx1250) — matrix operand order repaired and checked by reading, not compiled, not run on hardware
//
#include <hip/hip_runtime.h>

#define F_IN  512
#define F_HID 256
#define F_OUT 32

typedef __bf16 bf16_t;
typedef __attribute__((ext_vector_type(16))) __bf16 v16bf;
typedef __attribute__((ext_vector_type(8)))  __bf16 v8bf;
typedef __attribute__((ext_vector_type(8)))  float  v8f;
typedef __attribute__((ext_vector_type(4)))  float  v4f;

// ---------------- small helpers ----------------

__global__ void k_cvt_bf16(const float* __restrict__ in, bf16_t* __restrict__ out, int n) {
  int i = blockIdx.x * blockDim.x + threadIdx.x;
  if (i < n) out[i] = (bf16_t)in[i];
}

__global__ void k_deg_init(float* __restrict__ deg, int n) {
  int i = blockIdx.x * blockDim.x + threadIdx.x;
  if (i < n) deg[i] = 1.0f;  // self-loop contribution
}

__global__ void k_deg_edges(const long long* __restrict__ dst, float* __restrict__ deg, int E) {
  int e = blockIdx.x * blockDim.x + threadIdx.x;
  if (e < E) atomicAdd(&deg[(int)dst[e]], 1.0f);
}

__global__ void k_dinv(float* __restrict__ deg, int n) {
  int i = blockIdx.x * blockDim.x + threadIdx.x;
  if (i < n) {
    float d = deg[i];
    deg[i] = (d > 0.0f) ? rsqrtf(d) : 0.0f;  // in place: deg -> dinv
  }
}

__global__ void k_norm(const long long* __restrict__ src, const long long* __restrict__ dst,
                       const float* __restrict__ dinv, float* __restrict__ norm, int E) {
  int e = blockIdx.x * blockDim.x + threadIdx.x;
  if (e < E) norm[e] = dinv[(int)src[e]] * dinv[(int)dst[e]];
}

// assemble v16bf B fragment from two pre-swizzled 16B LDS chunks (h0 at p, h1 at p+256)
__device__ __forceinline__ v16bf load_b_frag(const bf16_t* __restrict__ p) {
  v8bf blo = *(const v8bf*)(p);
  v8bf bhi = *(const v8bf*)(p + 256);
  v16bf b;
#pragma unroll
  for (int i = 0; i < 8; ++i) { b[i] = blo[i]; b[i + 8] = bhi[i]; }
  return b;
}

// ---------------- GEMM1: h1 = relu(x @ W1^T + b1), bf16 WMMA ----------------
// Block = 256 threads = 8 waves; each wave computes a 16(M) x 256(N) strip.
// W1 K-slice (256 rows x 32 K = 16KB bf16) is staged in LDS (double buffered,
// pre-swizzled to fragment order) and shared by all 8 waves -> 8x less L2 traffic.
// x is streamed once from HBM as fp32, converted to bf16 in registers, with the
// next K-step's loads issued before the tile loop so they overlap the 16 WMMAs.

__global__ void __launch_bounds__(256)
k_mlp1(const float* __restrict__ x, const bf16_t* __restrict__ w1,
       const float* __restrict__ b1, bf16_t* __restrict__ h1, int M) {
  // 2 buffers x 1024 chunks x 8 bf16 (16B) = 32KB
  __shared__ __align__(16) bf16_t smem[2 * 8192];

  const int tid  = threadIdx.x;
  const int lane = tid & 31;
  const int wave = tid >> 5;
  const int m0   = blockIdx.x * 128 + wave * 16;

  const int mrow   = m0 + (lane & 15);
  const int mclamp = mrow < M ? mrow : (M - 1);
  const int klo    = (lane & 16) ? 8 : 0;    // A-fragment K sub-chunk select

  // fill of one LDS buffer: 1024 chunks of 16B, 4 per thread.
  // chunk id = t*64 + h*32 + l  ->  holds w1[(t*16+(l&15))*512 + kk + ((l&16)?8:0) + h*16 ..+7]
  auto fill_w1 = [&](int p, int kk) {
#pragma unroll
    for (int j = 0; j < 4; ++j) {
      int cid = tid + j * 256;
      int t   = cid >> 6;
      int rem = cid & 63;
      int h   = rem >> 5;
      int l   = rem & 31;
      int row = t * 16 + (l & 15);
      int kc  = kk + ((l & 16) ? 8 : 0) + h * 16;
      v8bf v  = *(const v8bf*)(w1 + (size_t)row * F_IN + kc);
      *(v8bf*)(&smem[p * 8192 + cid * 8]) = v;
    }
  };

  v8f acc[16];
#pragma unroll
  for (int t = 0; t < 16; ++t) acc[t] = (v8f){};

  const float* xrow = x + (size_t)mclamp * F_IN;

  // preload x chunks for kk = 0
  const float* xp = xrow + klo;
  v4f xc0 = *(const v4f*)(xp);
  v4f xc1 = *(const v4f*)(xp + 4);
  v4f xc2 = *(const v4f*)(xp + 16);
  v4f xc3 = *(const v4f*)(xp + 20);

  fill_w1(0, 0);
  __syncthreads();

  for (int kit = 0; kit < 16; ++kit) {
    const int kk  = kit * 32;
    const int cur = kit & 1;

    // start refilling the other LDS buffer with the next K-slice
    if (kit < 15) fill_w1(cur ^ 1, kk + 32);

    // ---- A fragment: convert fp32 -> bf16 in registers ----
    v16bf a;
#pragma unroll
    for (int i = 0; i < 4; ++i) {
      a[i]      = (bf16_t)xc0[i];
      a[i + 4]  = (bf16_t)xc1[i];
      a[i + 8]  = (bf16_t)xc2[i];
      a[i + 12] = (bf16_t)xc3[i];
    }
    // prefetch next K-step's x chunks (overlaps with the 16 WMMAs below)
    if (kit < 15) {
      const float* xn = xrow + kk + 32 + klo;
      xc0 = *(const v4f*)(xn);
      xc1 = *(const v4f*)(xn + 4);
      xc2 = *(const v4f*)(xn + 16);
      xc3 = *(const v4f*)(xn + 20);
    }

    // ---- 16 N-tiles; B fragments double-buffered from LDS ----
    const bf16_t* bbase = &smem[cur * 8192 + lane * 8];
    v16bf bc = load_b_frag(bbase);  // tile 0
#pragma unroll
    for (int t = 0; t < 16; ++t) {
      v16bf bn;
      if (t < 15) bn = load_b_frag(bbase + (t + 1) * 512);
      acc[t] = __builtin_amdgcn_wmma_f32_16x16x32_bf16(
          false, a, false, bc, (short)0, acc[t], false, false);
      bc = bn;
    }

    __syncthreads();
  }

  // ---- epilogue: bias + ReLU, store bf16 ----
  const int ncol  = lane & 15;
  const int rbase = m0 + ((lane >> 4) << 3);
#pragma unroll
  for (int t = 0; t < 16; ++t) {
    int col    = t * 16 + ncol;
    float bias = b1[col];
#pragma unroll
    for (int r = 0; r < 8; ++r) {
      int row = rbase + r;
      if (row < M) {
        float v = acc[t][r] + bias;
        v = v > 0.0f ? v : 0.0f;
        h1[(size_t)row * F_HID + col] = (bf16_t)v;
      }
    }
  }
}

// ---------------- GEMM2: h = h1 @ W2^T + b2 ; hidden = temp[0]*h ----------------

__global__ void __launch_bounds__(256)
k_mlp2(const bf16_t* __restrict__ h1, const bf16_t* __restrict__ w2,
       const float* __restrict__ b2, const float* __restrict__ temp,
       float* __restrict__ hcur, float* __restrict__ hidden, int M) {
  const int lane = threadIdx.x & 31;
  const int wave = threadIdx.x >> 5;
  const int m0   = blockIdx.x * 128 + wave * 16;

  const int mrow   = m0 + (lane & 15);
  const int mclamp = mrow < M ? mrow : (M - 1);
  const int klo    = (lane & 16) ? 8 : 0;
  const int kboff  = (lane & 16) ? 8 : 0;
  const int ncol   = lane & 15;

  v8f acc[2];
  acc[0] = (v8f){};
  acc[1] = (v8f){};

  const bf16_t* arow = h1 + (size_t)mclamp * F_HID;

  for (int kk = 0; kk < F_HID; kk += 32) {
    const bf16_t* ap = arow + kk + klo;
    v8bf a0 = *(const v8bf*)(ap);
    v8bf a1 = *(const v8bf*)(ap + 16);
    v16bf a;
#pragma unroll
    for (int i = 0; i < 8; ++i) { a[i] = a0[i]; a[i + 8] = a1[i]; }
#pragma unroll
    for (int t = 0; t < 2; ++t) {
      const bf16_t* wp = w2 + (size_t)(t * 16 + ncol) * F_HID + kk + kboff;
      v8bf blo = *(const v8bf*)(wp);
      v8bf bhi = *(const v8bf*)(wp + 16);
      v16bf b;
#pragma unroll
      for (int i = 0; i < 8; ++i) { b[i] = blo[i]; b[i + 8] = bhi[i]; }
      acc[t] = __builtin_amdgcn_wmma_f32_16x16x32_bf16(
          false, a, false, b, (short)0, acc[t], false, false);
    }
  }

  const float t0 = temp[0];
  const int rbase = m0 + ((lane >> 4) << 3);
#pragma unroll
  for (int t = 0; t < 2; ++t) {
    int col    = t * 16 + ncol;
    float bias = b2[col];
#pragma unroll
    for (int r = 0; r < 8; ++r) {
      int row = rbase + r;
      if (row < M) {
        float v = acc[t][r] + bias;
        hcur[(size_t)row * F_OUT + col]   = v;
        hidden[(size_t)row * F_OUT + col] = t0 * v;
      }
    }
  }
}

// ---------------- propagation ----------------

// hnext = dinv^2 * hcur   (self-loop term, also zero-initializes hnext)
__global__ void k_prop_init(float* __restrict__ hnext, const float* __restrict__ hcur,
                            const float* __restrict__ dinv, int n /* = M*32 */) {
  int i = blockIdx.x * blockDim.x + threadIdx.x;
  if (i < n) {
    float di = dinv[i >> 5];
    hnext[i] = di * di * hcur[i];
  }
}

// hnext[dst] += norm * hcur[src] ; 8 lanes x float4 per edge
__global__ void k_spmm(const long long* __restrict__ src, const long long* __restrict__ dst,
                       const float* __restrict__ norm, const float* __restrict__ hin,
                       float* __restrict__ hout, int E) {
  int tid = blockIdx.x * blockDim.x + threadIdx.x;
  int e = tid >> 3;
  if (e >= E) return;
  int c = (tid & 7) << 2;
  int s = (int)src[e];
  int d = (int)dst[e];
  float w = norm[e];
  v4f v = *(const v4f*)(hin + (size_t)s * F_OUT + c);
  float* o = hout + (size_t)d * F_OUT + c;
  atomicAdd(o + 0, w * v[0]);
  atomicAdd(o + 1, w * v[1]);
  atomicAdd(o + 2, w * v[2]);
  atomicAdd(o + 3, w * v[3]);
}

// out += temp[k] * h
__global__ void k_axpy(float* __restrict__ out, const float* __restrict__ h,
                       const float* __restrict__ temp, int k, int n) {
  int i = blockIdx.x * blockDim.x + threadIdx.x;
  if (i < n) out[i] += temp[k] * h[i];
}

// ---------------- launcher ----------------

extern "C" void kernel_launch(void* const* d_in, const int* in_sizes, int n_in,
                              void* d_out, int out_size, void* d_ws, size_t ws_size,
                              hipStream_t stream) {
  const float*     x    = (const float*)d_in[0];
  const long long* ei   = (const long long*)d_in[1];
  const float*     W1   = (const float*)d_in[2];
  const float*     b1   = (const float*)d_in[3];
  const float*     W2   = (const float*)d_in[4];
  const float*     b2   = (const float*)d_in[5];
  const float*     temp = (const float*)d_in[6];

  const int M = in_sizes[0] / F_IN;   // 100000
  const int E = in_sizes[1] / 2;      // 1600000
  const long long* src = ei;
  const long long* dst = ei + E;

  // workspace carve-out (256B aligned)
  char* ws = (char*)d_ws;
  size_t off = 0;
  auto carve = [&](size_t bytes) -> char* {
    char* p = ws + off;
    off += (bytes + 255) & ~(size_t)255;
    return p;
  };
  bf16_t* w1bf = (bf16_t*)carve((size_t)F_HID * F_IN * sizeof(bf16_t));
  bf16_t* w2bf = (bf16_t*)carve((size_t)F_OUT * F_HID * sizeof(bf16_t));
  bf16_t* h1bf = (bf16_t*)carve((size_t)M * F_HID * sizeof(bf16_t));
  float*  deg  = (float*)carve((size_t)M * sizeof(float));   // becomes dinv in place
  float*  norm = (float*)carve((size_t)E * sizeof(float));
  float*  hA   = (float*)carve((size_t)M * F_OUT * sizeof(float));
  float*  hB   = (float*)carve((size_t)M * F_OUT * sizeof(float));

  const int T = 256;
  auto cdiv = [](long long a, long long b) { return (int)((a + b - 1) / b); };

  // weights -> bf16
  k_cvt_bf16<<<cdiv(F_HID * F_IN, T), T, 0, stream>>>(W1, w1bf, F_HID * F_IN);
  k_cvt_bf16<<<cdiv(F_OUT * F_HID, T), T, 0, stream>>>(W2, w2bf, F_OUT * F_HID);

  // gcn_norm
  k_deg_init <<<cdiv(M, T), T, 0, stream>>>(deg, M);
  k_deg_edges<<<cdiv(E, T), T, 0, stream>>>(dst, deg, E);
  k_dinv     <<<cdiv(M, T), T, 0, stream>>>(deg, M);
  k_norm     <<<cdiv(E, T), T, 0, stream>>>(src, dst, deg, norm, E);

  // MLP (WMMA bf16)
  k_mlp1<<<cdiv(M, 128), 256, 0, stream>>>(x, w1bf, b1, h1bf, M);
  k_mlp2<<<cdiv(M, 128), 256, 0, stream>>>(h1bf, w2bf, b2, temp, hA, (float*)d_out, M);

  // GPR propagation, 10 hops
  float* cur = hA;
  float* nxt = hB;
  for (int k = 0; k < 10; ++k) {
    k_prop_init<<<cdiv((long long)M * F_OUT, T), T, 0, stream>>>(nxt, cur, deg, M * F_OUT);
    k_spmm     <<<cdiv((long long)E * 8, T), T, 0, stream>>>(src, dst, norm, cur, nxt, E);
    k_axpy     <<<cdiv((long long)M * F_OUT, T), T, 0, stream>>>((float*)d_out, nxt, temp, k + 1, M * F_OUT);
    float* t = cur; cur = nxt; nxt = t;
  }
}
